// MambaNoisePredictor_70995809402932
// MI455X (gfx1250) — compile-verified, matrix operand order checked by reading, not run on hardware
//
#include <hip/hip_runtime.h>
#include <hip/hip_bf16.h>
#include <stdint.h>

// ---------------------------------------------------------------------------
// MI455X (gfx1250) implementation of the Mamba noise-predictor reference.
// All dense projections run on v_wmma_f32_16x16x32_bf16 (fp32 accumulate),
// 64x64 output tile per wave (16 WMMAs per 8 fragment loads -> 2 WMMA/load).
// The sequential SSM scan uses wave32 lane-parallel states + shfl reductions.
// ---------------------------------------------------------------------------

typedef __bf16 bf16x16 __attribute__((ext_vector_type(16)));
typedef float  f32x8   __attribute__((ext_vector_type(8)));

union FragU {
  bf16x16 v;
  uint4   q[2];
};

__device__ __forceinline__ unsigned short f2bf(float f) {
  unsigned int u = __float_as_uint(f);
  unsigned int r = u + 0x7FFFu + ((u >> 16) & 1u);   // round-to-nearest-even
  return (unsigned short)(r >> 16);
}

__device__ __forceinline__ float gelu_exact(float x) {
  return 0.5f * x * (1.f + erff(x * 0.70710678118654752f));
}
__device__ __forceinline__ float softplus_f(float x) {
  return (x > 20.f) ? x : log1pf(__expf(x));
}
__device__ __forceinline__ float silu_f(float x) {
  return x / (1.f + __expf(-x));
}

// Load one 16x32 bf16 fragment (A or B operand) from a row-major tile.
// ISA layout (16-bit A 16x32): lanes 0-15 -> rows 0-15 with K in
// {0..7}∪{16..23}; lanes 16-31 -> rows 0-15 with K in {8..15}∪{24..31}.
__device__ __forceinline__ bf16x16 load_frag(const unsigned short* __restrict__ base,
                                             int ld, int lane) {
  const unsigned short* p = base + (size_t)(lane & 15) * ld + ((lane >> 4) << 3);
  FragU f;
  f.q[0] = *reinterpret_cast<const uint4*>(p);        // K = kb..kb+7
  f.q[1] = *reinterpret_cast<const uint4*>(p + 16);   // K = kb+16..kb+23
  return f.v;
}

// ---------------------------------------------------------------------------
// Generic bf16 WMMA GEMM: C[M,N] = A[M,K] * W[N,K]^T (+bias +addsrc, act)
// One wave computes a 64x64 output tile (4x4 WMMA accumulators).
// Requires M % 64 == 0, N % 64 == 0, K % 32 == 0 (true for every call here).
// act: 0=none, 1=softplus. Optional bf16 mirror output Cbf.
// ---------------------------------------------------------------------------
__global__ void __launch_bounds__(256)
gemm_bf16_wmma(const unsigned short* __restrict__ A,
               const unsigned short* __restrict__ W,
               float* __restrict__ C,
               unsigned short* __restrict__ Cbf,
               const float* __restrict__ bias,
               const float* __restrict__ addsrc,
               int M, int N, int K, int ldA, int ldW, int ldC, int act)
{
  const int lane   = threadIdx.x & 31;
  const int wave   = blockIdx.x * (blockDim.x >> 5) + (threadIdx.x >> 5);
  const int ntiles = N >> 6;
  const int nwaves = (M >> 6) * ntiles;
  if (wave >= nwaves) return;               // wave-uniform: EXEC stays all-1s
  const int mt = wave / ntiles;
  const int nt = wave - mt * ntiles;

  const unsigned short* Abase = A + (size_t)(mt << 6) * ldA;
  const unsigned short* Wbase = W + (size_t)(nt << 6) * ldW;

  f32x8 acc[4][4] = {};

  for (int kb = 0; kb < K; kb += 32) {
    bf16x16 af[4], bfr[4];
#pragma unroll
    for (int i = 0; i < 4; ++i)
      af[i] = load_frag(Abase + (size_t)(i << 4) * ldA + kb, ldA, lane);
#pragma unroll
    for (int j = 0; j < 4; ++j)
      bfr[j] = load_frag(Wbase + (size_t)(j << 4) * ldW + kb, ldW, lane);
    if (kb + 32 < K) {   // -> global_prefetch_b8 (near-cache) for next K slice
      __builtin_prefetch(Abase + (size_t)(lane & 15) * ldA + kb + 32, 0, 3);
      __builtin_prefetch(Wbase + (size_t)(lane & 15) * ldW + kb + 32, 0, 3);
    }
#pragma unroll
    for (int i = 0; i < 4; ++i)
#pragma unroll
      for (int j = 0; j < 4; ++j)
        acc[i][j] = __builtin_amdgcn_wmma_f32_16x16x32_bf16(
            false, af[i], false, bfr[j], (short)0, acc[i][j], false, false);
  }

  // C/D layout: VGPR v holds row v (lanes 0-15) or row v+8 (lanes 16-31),
  // column = lane & 15.
  const int colA = lane & 15;
  const int rb   = (lane >> 4) << 3;

#pragma unroll
  for (int ti = 0; ti < 4; ++ti) {
#pragma unroll
    for (int tj = 0; tj < 4; ++tj) {
      const int row0 = (mt << 6) + ti * 16 + rb;
      const int col  = (nt << 6) + tj * 16 + colA;
      const float bv = bias ? bias[col] : 0.f;
#pragma unroll
      for (int v = 0; v < 8; ++v) {
        const size_t o = (size_t)(row0 + v) * ldC + col;
        float val = acc[ti][tj][v] + bv;
        if (addsrc) val += addsrc[o];
        if (act == 1) val = softplus_f(val);
        if (C)   C[o]   = val;
        if (Cbf) Cbf[o] = f2bf(val);
      }
    }
  }
}

// ---------------------------------------------------------------------------
// RMSNorm per row (block per row). apply_gelu: 0 -> bf16 out, 1 -> f32 out.
// ---------------------------------------------------------------------------
__global__ void __launch_bounds__(256)
rmsnorm_kernel(const float* __restrict__ in, const float* __restrict__ w,
               float* __restrict__ out_f32, unsigned short* __restrict__ out_bf,
               int D, int apply_gelu)
{
  const int row = blockIdx.x;
  const float* r = in + (size_t)row * D;
  float s = 0.f;
  for (int i = threadIdx.x; i < D; i += blockDim.x) { float v = r[i]; s += v * v; }
  for (int off = 16; off > 0; off >>= 1) s += __shfl_xor(s, off, 32);
  __shared__ float ls[8];
  const int wid = threadIdx.x >> 5;
  if ((threadIdx.x & 31) == 0) ls[wid] = s;
  __syncthreads();
  if (threadIdx.x == 0) {
    float t = 0.f;
    for (int i = 0; i < (int)(blockDim.x >> 5); ++i) t += ls[i];
    ls[0] = t;
  }
  __syncthreads();
  const float rr = rsqrtf(ls[0] / (float)D + 1e-5f);
  for (int i = threadIdx.x; i < D; i += blockDim.x) {
    float v = r[i] * rr * w[i];
    if (apply_gelu) v = gelu_exact(v);
    if (out_f32) out_f32[(size_t)row * D + i] = v;
    if (out_bf)  out_bf[(size_t)row * D + i]  = f2bf(v);
  }
}

// ---------------------------------------------------------------------------
// Depthwise causal conv (K=4) + bias + SiLU over u = xz[:, :Di].
// ---------------------------------------------------------------------------
__global__ void conv_silu_kernel(const float* __restrict__ xz,
                                 const float* __restrict__ cw,
                                 const float* __restrict__ cb,
                                 float* __restrict__ ucv,
                                 unsigned short* __restrict__ ubf,
                                 int M, int L, int Di)
{
  int idx = blockIdx.x * blockDim.x + threadIdx.x;
  if (idx >= M * Di) return;
  int d = idx % Di;
  int m = idx / Di;
  int l = m % L;
  float acc = cb[d];
#pragma unroll
  for (int k = 0; k < 4; ++k) {
    int lt = l - 3 + k;
    if (lt >= 0) acc += xz[(size_t)(m - 3 + k) * (2 * Di) + d] * cw[d * 4 + k];
  }
  float s = silu_f(acc);
  ucv[idx] = s;
  ubf[idx] = f2bf(s);
}

// ---------------------------------------------------------------------------
// Selective scan. Each wave: 2 channels; 16 lanes = 16 SSM states/channel.
// y[b,t,d] = sum_n h_n * C[t,n] + u*D, via shfl_xor tree reduction (wave32).
// ---------------------------------------------------------------------------
__global__ void __launch_bounds__(256)
scan_kernel(const float* __restrict__ dt, const float* __restrict__ u,
            const float* __restrict__ xdbl, const float* __restrict__ Alog,
            const float* __restrict__ Dp, float* __restrict__ ys,
            int Bsz, int L, int Di)
{
  const int lane = threadIdx.x & 31;
  const int wave = blockIdx.x * (blockDim.x >> 5) + (threadIdx.x >> 5);
  const int n    = lane & 15;
  const int ch   = (wave << 1) + (lane >> 4);
  if (ch >= Bsz * Di) return;
  const int b = ch / Di;
  const int d = ch - b * Di;

  const float Acoef = -__expf(Alog[d * 16 + n]);
  const float Dd    = Dp[d];
  const float* dtp = dt   + (size_t)(b * L) * Di + d;
  const float* up  = u    + (size_t)(b * L) * Di + d;
  const float* xp  = xdbl + (size_t)(b * L) * 64 + 32 + n;   // B at +32, C at +48
  float* yp = ys + (size_t)(b * L) * Di + d;

  float hstate = 0.f;
  for (int t = 0; t < L; ++t) {
    const float dtt = dtp[(size_t)t * Di];
    const float ut  = up[(size_t)t * Di];
    const float Bt  = xp[(size_t)t * 64];
    const float Ct  = xp[(size_t)t * 64 + 16];
    hstate = __expf(dtt * Acoef) * hstate + (dtt * ut) * Bt;
    float part = hstate * Ct;
    part += __shfl_xor(part, 1, 32);
    part += __shfl_xor(part, 2, 32);
    part += __shfl_xor(part, 4, 32);
    part += __shfl_xor(part, 8, 32);
    if (n == 0) yp[(size_t)t * Di] = part + ut * Dd;
  }
}

// y = (scan_y) * silu(z); z = xz[:, Di:2Di]; emit bf16 for out_proj GEMM.
__global__ void ysilu_kernel(const float* __restrict__ ys,
                             const float* __restrict__ xz,
                             unsigned short* __restrict__ ybf, int M, int Di)
{
  int idx = blockIdx.x * blockDim.x + threadIdx.x;
  if (idx >= M * Di) return;
  int d = idx % Di;
  int m = idx / Di;
  float z = xz[(size_t)m * (2 * Di) + Di + d];
  ybf[idx] = f2bf(ys[idx] * silu_f(z));
}

// im2col for the 3-tap temporal conv: A3[m, j*Dm+d] = h[m+j-1, d] (0 at edges).
__global__ void im2col3_kernel(const float* __restrict__ h,
                               unsigned short* __restrict__ a3,
                               int M, int L, int Dm)
{
  int idx = blockIdx.x * blockDim.x + threadIdx.x;
  int K3 = 3 * Dm;
  if (idx >= M * K3) return;
  int m = idx / K3;
  int k = idx - m * K3;
  int j = k / Dm;
  int d = k - j * Dm;
  int l = m % L;
  int lt = l + j - 1;
  float v = (lt >= 0 && lt < L) ? h[(size_t)(m + j - 1) * Dm + d] : 0.f;
  a3[idx] = f2bf(v);
}

// tc_w (H, Dm, 3) -> bf16 W[H, j*Dm+d] for the im2col GEMM.
__global__ void tcw_permute_kernel(const float* __restrict__ tcw,
                                   unsigned short* __restrict__ out,
                                   int H, int Dm)
{
  int idx = blockIdx.x * blockDim.x + threadIdx.x;
  int K3 = 3 * Dm;
  if (idx >= H * K3) return;
  int ch = idx / K3;
  int k  = idx - ch * K3;
  int j  = k / Dm;
  int d  = k - j * Dm;
  out[idx] = f2bf(tcw[(size_t)ch * K3 + d * 3 + j]);
}

// BatchNorm stats over (B, L) per channel: stat[ch]=mu, stat[H+ch]=rstd.
__global__ void __launch_bounds__(256)
bn_stats_kernel(const float* __restrict__ hc, float* __restrict__ stat, int M, int H)
{
  const int ch = blockIdx.x;
  float s = 0.f, s2 = 0.f;
  for (int m = threadIdx.x; m < M; m += blockDim.x) {
    float v = hc[(size_t)m * H + ch];
    s += v; s2 += v * v;
  }
  for (int off = 16; off > 0; off >>= 1) {
    s  += __shfl_xor(s,  off, 32);
    s2 += __shfl_xor(s2, off, 32);
  }
  __shared__ float l1[8], l2[8];
  int wid = threadIdx.x >> 5;
  if ((threadIdx.x & 31) == 0) { l1[wid] = s; l2[wid] = s2; }
  __syncthreads();
  if (threadIdx.x == 0) {
    float t1 = 0.f, t2 = 0.f;
    for (int i = 0; i < 8; ++i) { t1 += l1[i]; t2 += l2[i]; }
    float mu  = t1 / (float)M;
    float var = t2 / (float)M - mu * mu;
    stat[ch]     = mu;
    stat[H + ch] = rsqrtf(var + 1e-5f);
  }
}

__global__ void bn_gelu_kernel(const float* __restrict__ hc,
                               const float* __restrict__ stat,
                               const float* __restrict__ g,
                               const float* __restrict__ bb,
                               unsigned short* __restrict__ hcbf, int M, int H)
{
  int idx = blockIdx.x * blockDim.x + threadIdx.x;
  if (idx >= M * H) return;
  int ch = idx % H;
  float v = (hc[idx] - stat[ch]) * stat[H + ch] * g[ch] + bb[ch];
  hcbf[idx] = f2bf(gelu_exact(v));
}

__global__ void convert_f32_bf16_kernel(const float* __restrict__ in,
                                        unsigned short* __restrict__ out, int n)
{
  int i = blockIdx.x * blockDim.x + threadIdx.x;
  if (i < n) out[i] = f2bf(in[i]);
}

// ---------------------------------------------------------------------------
// Host orchestration
// ---------------------------------------------------------------------------
extern "C" void kernel_launch(void* const* d_in, const int* in_sizes, int n_in,
                              void* d_out, int out_size, void* d_ws, size_t ws_size,
                              hipStream_t stream)
{
  (void)in_sizes; (void)n_in; (void)out_size; (void)ws_size;

  const float* x        = (const float*)d_in[0];
  const float* ip_w     = (const float*)d_in[1];
  const float* ip_b     = (const float*)d_in[2];
  const float* ip_norm  = (const float*)d_in[3];
  const float* blk_norm = (const float*)d_in[4];
  const float* in_w     = (const float*)d_in[5];
  const float* conv_w   = (const float*)d_in[6];
  const float* conv_b   = (const float*)d_in[7];
  const float* x_w      = (const float*)d_in[8];
  const float* dt_w     = (const float*)d_in[9];
  const float* dt_b     = (const float*)d_in[10];
  const float* A_log    = (const float*)d_in[11];
  const float* Dp       = (const float*)d_in[12];
  const float* out_w    = (const float*)d_in[13];
  const float* tc_w     = (const float*)d_in[14];
  const float* tc_b     = (const float*)d_in[15];
  const float* bn_g     = (const float*)d_in[16];
  const float* bn_b     = (const float*)d_in[17];
  const float* fp_w     = (const float*)d_in[18];
  const float* fp_b     = (const float*)d_in[19];
  float* out = (float*)d_out;

  constexpr int NL = 2, Bsz = 2, L = 2048, M = Bsz * L;
  constexpr int Dm = 512, Di = 1024, H = 1024, Dd = 256, K3 = 3 * Dm;

  // ---- workspace layout (fixed, deterministic) ----
  char* ws = (char*)d_ws;
  size_t off = 0;
  auto alloc = [&](size_t bytes) -> char* {
    char* p = ws + off;
    off = (off + bytes + 255) & ~(size_t)255;
    return p;
  };
  float*          h      = (float*)alloc((size_t)M * Dm * 4);       // residual
  unsigned short* hnbf   = (unsigned short*)alloc((size_t)M * Dm * 2); // x_bf / hn_bf
  float*          xz     = (float*)alloc((size_t)M * 2 * Di * 4);
  float*          ucv    = (float*)alloc((size_t)M * Di * 4);
  unsigned short* ubf    = (unsigned short*)alloc((size_t)M * Di * 2);
  float*          xdbl   = (float*)alloc((size_t)M * 64 * 4);
  unsigned short* xdblbf = (unsigned short*)alloc((size_t)M * 64 * 2);
  float*          dt     = (float*)alloc((size_t)M * Di * 4);
  float*          ys     = (float*)alloc((size_t)M * Di * 4);
  unsigned short* ybf    = (unsigned short*)alloc((size_t)M * Di * 2);
  unsigned short* a3     = (unsigned short*)alloc((size_t)M * K3 * 2);
  float*          hcf    = (float*)alloc((size_t)M * H * 4);        // also ip GEMM scratch
  unsigned short* hcbf   = (unsigned short*)alloc((size_t)M * H * 2);
  float*          stat   = (float*)alloc((size_t)2 * H * 4);
  unsigned short* ipw_bf = (unsigned short*)alloc((size_t)Dm * Dm * 2);
  unsigned short* inw_bf = (unsigned short*)alloc((size_t)NL * 2 * Di * Dm * 2);
  unsigned short* xw_bf  = (unsigned short*)alloc((size_t)NL * 64 * Di * 2);
  unsigned short* dtw_bf = (unsigned short*)alloc((size_t)NL * Di * 32 * 2);
  unsigned short* ow_bf  = (unsigned short*)alloc((size_t)NL * Dm * Di * 2);
  unsigned short* tcw_bf = (unsigned short*)alloc((size_t)H * K3 * 2);
  unsigned short* fpw_bf = (unsigned short*)alloc((size_t)Dd * Di * 2);

  auto cvt = [&](const float* src, unsigned short* dst, int n) {
    convert_f32_bf16_kernel<<<(n + 255) / 256, 256, 0, stream>>>(src, dst, n);
  };
  auto gemm = [&](const unsigned short* A, int ldA, const unsigned short* W, int ldW,
                  float* C, unsigned short* Cbf, int ldC,
                  const float* bias, const float* addsrc,
                  int Mi, int Ni, int Ki, int act) {
    int waves  = (Mi >> 6) * (Ni >> 6);
    int blocks = (waves * 32 + 255) / 256;
    gemm_bf16_wmma<<<blocks, 256, 0, stream>>>(A, W, C, Cbf, bias, addsrc,
                                               Mi, Ni, Ki, ldA, ldW, ldC, act);
  };

  // ---- one-time (per launch) bf16 weight conversion ----
  cvt(x,     hnbf,   M * Dm);                 // consumed by ip GEMM before reuse
  cvt(ip_w,  ipw_bf, Dm * Dm);
  cvt(in_w,  inw_bf, NL * 2 * Di * Dm);
  cvt(x_w,   xw_bf,  NL * 64 * Di);
  cvt(dt_w,  dtw_bf, NL * Di * 32);
  cvt(out_w, ow_bf,  NL * Dm * Di);
  cvt(fp_w,  fpw_bf, Dd * Di);
  tcw_permute_kernel<<<(H * K3 + 255) / 256, 256, 0, stream>>>(tc_w, tcw_bf, H, Dm);

  // ---- input projection + RMSNorm + GELU ----
  gemm(hnbf, Dm, ipw_bf, Dm, hcf, nullptr, Dm, ip_b, nullptr, M, Dm, Dm, 0);
  rmsnorm_kernel<<<M, 256, 0, stream>>>(hcf, ip_norm, h, nullptr, Dm, 1);

  // ---- Mamba blocks ----
  for (int i = 0; i < NL; ++i) {
    rmsnorm_kernel<<<M, 256, 0, stream>>>(h, blk_norm + i * Dm, nullptr, hnbf, Dm, 0);
    // xz = hn @ in_proj^T  (M=4096, N=2048, K=512)
    gemm(hnbf, Dm, inw_bf + (size_t)i * 2 * Di * Dm, Dm, xz, nullptr, 2 * Di,
         nullptr, nullptr, M, 2 * Di, Dm, 0);
    // depthwise causal conv + SiLU
    conv_silu_kernel<<<(M * Di + 255) / 256, 256, 0, stream>>>(
        xz, conv_w + (size_t)i * Di * 4, conv_b + (size_t)i * Di, ucv, ubf, M, L, Di);
    // xdbl = u @ x_proj^T  (N=64)
    gemm(ubf, Di, xw_bf + (size_t)i * 64 * Di, Di, xdbl, xdblbf, 64,
         nullptr, nullptr, M, 64, Di, 0);
    // dt = softplus(xdbl[:, :32] @ dt_proj^T + dtb)  (K=32)
    gemm(xdblbf, 64, dtw_bf + (size_t)i * Di * 32, 32, dt, nullptr, Di,
         dt_b + (size_t)i * Di, nullptr, M, Di, 32, 1);
    // sequential selective scan
    scan_kernel<<<(Bsz * Di / 2 * 32 + 255) / 256, 256, 0, stream>>>(
        dt, ucv, xdbl, A_log + (size_t)i * Di * 16, Dp + (size_t)i * Di, ys, Bsz, L, Di);
    // y = (scan + u*D) * silu(z)
    ysilu_kernel<<<(M * Di + 255) / 256, 256, 0, stream>>>(ys, xz, ybf, M, Di);
    // h = h + y @ out_proj^T
    gemm(ybf, Di, ow_bf + (size_t)i * Dm * Di, Di, h, nullptr, Dm,
         nullptr, h, M, Dm, Di, 0);
  }

  // ---- temporal conv (as im2col GEMM, K=1536) + BN + GELU ----
  im2col3_kernel<<<(M * K3 + 255) / 256, 256, 0, stream>>>(h, a3, M, L, Dm);
  gemm(a3, K3, tcw_bf, K3, hcf, nullptr, H, tc_b, nullptr, M, H, K3, 0);
  bn_stats_kernel<<<H, 256, 0, stream>>>(hcf, stat, M, H);
  bn_gelu_kernel<<<(M * H + 255) / 256, 256, 0, stream>>>(hcf, stat, bn_g, bn_b, hcbf, M, H);

  // ---- final projection ----
  gemm(hcbf, Di, fpw_bf, Di, out, nullptr, Dd, fp_b, nullptr, M, Dd, Di, 0);
}
